// Inputspace_42563125903929
// MI455X (gfx1250) — matrix operand order repaired and checked by reading, not compiled, not run on hardware
//
#include <hip/hip_runtime.h>

// ---------------- problem constants ----------------
#define HID   64
#define HEADS 4
#define DH    16        // HID / HEADS
#define BATCH 512
#define T     2
#define ND    8
#define NF    2048
#define KT    64        // key tile (4 M-tiles of 16 for WMMA)

// output layout (concatenated flat, return order)
constexpr size_t OUT_REPR   = 0;
constexpr size_t OUT_FIRE_E = (size_t)BATCH * ND * HID;                 // 262,144
constexpr size_t OUT_ATTN   = OUT_FIRE_E + (size_t)BATCH * NF * HID;    // + 67,108,864
constexpr size_t OUT_MASK   = OUT_ATTN + (size_t)BATCH * ND * NF;       // + 8,388,608

// input indices (setup_inputs insertion order, params dict in insertion order)
enum {
  IN_STATES = 0, IN_GOALS, IN_FIREPOS, IN_FIREMASK, IN_FIREGLOB,
  P_POS_W, P_POS_B, P_GOAL_W, P_GOAL_B, P_FIRE_W, P_FIRE_B, P_VEL_W, P_VEL_B,
  P_FC_W, P_FC_B, P_CEN_W, P_CEN_B, P_WIND_W, P_WIND_B,
  P_ID_EMB, P_FUS_W1, P_FUS_B1, P_FUS_W2, P_FUS_B2,
  P_SA_WQ, P_SA_BQ, P_SA_WK, P_SA_BK, P_SA_WV, P_SA_BV, P_SA_WO, P_SA_BO,
  P_SA_LNG, P_SA_LNB,
  P_CA_WQ, P_CA_BQ, P_CA_WK, P_CA_BK, P_CA_WV, P_CA_BV, P_CA_WO, P_CA_BO,
  P_CA_LNG, P_CA_LNB,
  N_IN
};

struct PtrPack { const void* p[N_IN]; };
#define PF(i) ((const float*)pp.p[(i)])

typedef __attribute__((ext_vector_type(16))) _Float16 v16h;
typedef __attribute__((ext_vector_type(8)))  float    v8f;

union H16 { v16h v; _Float16 h[16]; };

// ---------------- WMMA fragment gathers (CDNA5 ISA layouts) ---------
// A: 16x32 f16 tile (M x K) from row-major source [rows x ldm].
// Per-lane halves are two contiguous 16B runs -> compiler merges to 2x ds_load_b128.
__device__ __forceinline__ v16h load_a_frag(const _Float16* base, int mt, int ks,
                                            int lane, int ldm) {
  H16 u;
  const int M   = lane & 15;
  const int hiK = (lane >> 4) * 8;   // lanes 16-31: K group shifted by +8
  #pragma unroll
  for (int j = 0; j < 16; ++j) {
    const int vg = j >> 1, lo = j & 1;
    int K = (vg < 4) ? (vg * 2 + lo) : (16 + (vg - 4) * 2 + lo);
    K += hiK;
    u.h[j] = base[(mt * 16 + M) * ldm + ks * 32 + K];
  }
  return u.v;
}

// B operand staged in fragment-ready order: [ks][nt][lane][j] (16 halves
// contiguous per lane) -> one 32B v16h load (2x ds_load_b128), no packing VALU.
__device__ __forceinline__ v16h load_b_ready(const _Float16* fragBase, int ks,
                                             int nt, int lane) {
  return *(const v16h*)&fragBase[(((ks * 4 + nt) * 32) + lane) * 16];
}

// ---------------- kernel 0: effective key-padding mask -------------------------
__global__ void __launch_bounds__(256)
mask_kernel(const unsigned char* __restrict__ msk, float* __restrict__ out) {
  const int b = blockIdx.x;
  __shared__ int sAny;
  if (threadIdx.x == 0) sAny = 0;
  __syncthreads();
  int any = 0;
  for (int k = threadIdx.x; k < NF; k += 256) any |= (int)msk[(size_t)b * NF + k];
  if (any) atomicOr(&sAny, 1);
  __syncthreads();
  const int fully = !sAny;  // fully_masked = no valid fire anywhere
  float* om = out + OUT_MASK + (size_t)b * NF;
  for (int k = threadIdx.x; k < NF; k += 256) {
    int eff = msk[(size_t)b * NF + k] ? 0 : 1;   // mask = ~fire_positions_mask
    if (k == 0 && fully) eff = 0;                // un-mask key 0 if all masked
    om[k] = (float)eff;
  }
}

// ---------------- kernel 1: drone encoder (tiny, VALU) -------------------------
__global__ void __launch_bounds__(64)
drone_encoder(PtrPack pp, float* __restrict__ ws_demb, float* __restrict__ ws_q) {
  const int b = blockIdx.x;
  const int h = threadIdx.x;  // hidden column owned by this thread

  __shared__ float X[ND][4 * HID];
  __shared__ float H1[ND][2 * HID];
  __shared__ float E[ND][HID];
  __shared__ float Qs[ND][HID], Ks[ND][HID], Vs[ND][HID];
  __shared__ float SC[HEADS][ND][ND];
  __shared__ float Os[ND][HID];
  __shared__ float Ys[ND][HID];
  __shared__ float st[ND][2];

  const float* states = PF(IN_STATES) + (size_t)b * T * ND * 2;
  const float* goals  = PF(IN_GOALS)  + (size_t)b * ND * 2;
  const float* fg     = PF(IN_FIREGLOB) + (size_t)b * 5;

  for (int d = 0; d < ND; ++d) {
    const float px = states[(1 * ND + d) * 2 + 0], py = states[(1 * ND + d) * 2 + 1];
    const float ox = states[(0 * ND + d) * 2 + 0], oy = states[(0 * ND + d) * 2 + 1];
    const float vx = px - ox, vy = py - oy;
    const float gx = goals[d * 2 + 0], gy = goals[d * 2 + 1];
    X[d][h]           = fmaxf(px * PF(P_POS_W)[h]  + py * PF(P_POS_W)[HID + h]  + PF(P_POS_B)[h],  0.f);
    X[d][HID + h]     = fmaxf(gx * PF(P_GOAL_W)[h] + gy * PF(P_GOAL_W)[HID + h] + PF(P_GOAL_B)[h], 0.f);
    X[d][2 * HID + h] = PF(P_ID_EMB)[d * HID + h];
    X[d][3 * HID + h] = fmaxf(vx * PF(P_VEL_W)[h]  + vy * PF(P_VEL_W)[HID + h]  + PF(P_VEL_B)[h],  0.f);
  }
  __syncthreads();

  const float* W1 = PF(P_FUS_W1);           // (256,128)
  for (int d = 0; d < ND; ++d)
    for (int f = h; f < 2 * HID; f += 64) {
      float acc = PF(P_FUS_B1)[f];
      for (int k = 0; k < 4 * HID; ++k) acc += X[d][k] * W1[k * (2 * HID) + f];
      H1[d][f] = fmaxf(acc, 0.f);
    }
  __syncthreads();

  const float fcnt = fg[0], cx = fg[1], cy = fg[2], wx = fg[3], wy = fg[4];
  const float gctx =
      fmaxf(fcnt * PF(P_FC_W)[h] + PF(P_FC_B)[h], 0.f) +
      fmaxf(cx * PF(P_CEN_W)[h]  + cy * PF(P_CEN_W)[HID + h]  + PF(P_CEN_B)[h],  0.f) +
      fmaxf(wx * PF(P_WIND_W)[h] + wy * PF(P_WIND_W)[HID + h] + PF(P_WIND_B)[h], 0.f);

  const float* W2 = PF(P_FUS_W2);           // (128,64)
  for (int d = 0; d < ND; ++d) {
    float acc = PF(P_FUS_B2)[h];
    for (int k = 0; k < 2 * HID; ++k) acc += H1[d][k] * W2[k * HID + h];
    E[d][h] = fmaxf(acc, 0.f) + gctx;
  }
  __syncthreads();

  // self-attention QKV
  for (int d = 0; d < ND; ++d) {
    float aq = PF(P_SA_BQ)[h], ak = PF(P_SA_BK)[h], av = PF(P_SA_BV)[h];
    for (int k = 0; k < HID; ++k) {
      const float e = E[d][k];
      aq += e * PF(P_SA_WQ)[k * HID + h];
      ak += e * PF(P_SA_WK)[k * HID + h];
      av += e * PF(P_SA_WV)[k * HID + h];
    }
    Qs[d][h] = aq; Ks[d][h] = ak; Vs[d][h] = av;
  }
  __syncthreads();

  // scores (4 heads x 8 x 8), scale = 1/sqrt(16)
  for (int ii = 0; ii < 4; ++ii) {
    const int idx = h * 4 + ii;
    const int hh = idx >> 6, rem = idx & 63, qi = rem >> 3, kj = rem & 7;
    float s = 0.f;
    for (int d2 = 0; d2 < DH; ++d2) s += Qs[qi][hh * DH + d2] * Ks[kj][hh * DH + d2];
    SC[hh][qi][kj] = s * 0.25f;
  }
  __syncthreads();
  if (h < HEADS * ND) {
    const int hh = h >> 3, qi = h & 7;
    float mx = -1e30f;
    for (int j = 0; j < ND; ++j) mx = fmaxf(mx, SC[hh][qi][j]);
    float sum = 0.f;
    for (int j = 0; j < ND; ++j) { const float e = __expf(SC[hh][qi][j] - mx); SC[hh][qi][j] = e; sum += e; }
    const float inv = 1.f / sum;
    for (int j = 0; j < ND; ++j) SC[hh][qi][j] *= inv;
  }
  __syncthreads();
  {
    const int hh = h / DH;
    for (int i = 0; i < ND; ++i) {
      float acc = 0.f;
      for (int j = 0; j < ND; ++j) acc += SC[hh][i][j] * Vs[j][h];
      Os[i][h] = acc;
    }
  }
  __syncthreads();
  for (int i = 0; i < ND; ++i) {
    float acc = PF(P_SA_BO)[h];
    for (int d2 = 0; d2 < HID; ++d2) acc += Os[i][d2] * PF(P_SA_WO)[d2 * HID + h];
    Ys[i][h] = E[i][h] + acc;
  }
  __syncthreads();
  if (h < ND) {
    float m = 0.f;
    for (int c = 0; c < HID; ++c) m += Ys[h][c];
    m /= HID;
    float v = 0.f;
    for (int c = 0; c < HID; ++c) { const float d3 = Ys[h][c] - m; v += d3 * d3; }
    v /= HID;
    st[h][0] = m; st[h][1] = rsqrtf(v + 1e-5f);
  }
  __syncthreads();
  for (int i = 0; i < ND; ++i) {
    const float y = (Ys[i][h] - st[i][0]) * st[i][1] * PF(P_SA_LNG)[h] + PF(P_SA_LNB)[h];
    E[i][h] = y;
    ws_demb[((size_t)b * ND + i) * HID + h] = y;
  }
  __syncthreads();
  for (int i = 0; i < ND; ++i) {
    float acc = PF(P_CA_BQ)[h];
    for (int k = 0; k < HID; ++k) acc += E[i][k] * PF(P_CA_WQ)[k * HID + h];
    ws_q[((size_t)b * ND + i) * HID + h] = acc;
  }
}

// ---------------- kernel 2: fused fire embed + WMMA cross-attention ------------
__global__ void __launch_bounds__(256)
fire_cross_attn(PtrPack pp, const float* __restrict__ ws_demb,
                const float* __restrict__ ws_q, float* __restrict__ out) {
  const int b = blockIdx.x;
  const int tid = threadIdx.x;
  const int wave = tid >> 5, lane = tid & 31;

  // B operands stored pre-swizzled in WMMA fragment order [ks][nt][lane][16]
  __shared__ alignas(32) _Float16 sWkF[2 * 4 * 32 * 16];   // 4096 halves
  __shared__ alignas(32) _Float16 sWvF[2 * 4 * 32 * 16];
  __shared__ float    sFw[2 * HID], sFb[HID], sBk[HID], sBv[HID];
  __shared__ alignas(32) _Float16 sFe[KT * HID];      // fire_e tile (A matrix)
  __shared__ alignas(32) _Float16 sK16[KT * HID];     // projected K tile
  __shared__ alignas(32) _Float16 sV16[KT * HID];     // projected V tile
  __shared__ float    sQ[ND * HID];
  __shared__ float    sMask[KT];
  __shared__ float    sML[HEADS * ND * 2];
  __shared__ float    sO[ND * HID];
  __shared__ float    sY[ND * HID];
  __shared__ float    sStat[2 * ND];

  // stage weights (f32 -> f16) into LDS, directly in B-fragment order:
  // fragIdx = ((ks*4 + nt)*32 + lane)*16 + j ;  K = ks*32 + (j&7) + 16*(j>>3) + 8*(lane>>4),
  // N = nt*16 + (lane&15)  (CDNA5 ISA 16-bit B layout)
  {
    const float* Wk = PF(P_CA_WK);
    const float* Wv = PF(P_CA_WV);
    for (int idx = tid; idx < 2 * 4 * 32 * 16; idx += 256) {
      const int j  = idx & 15;
      const int ln = (idx >> 4) & 31;
      const int nt = (idx >> 9) & 3;
      const int ks = (idx >> 11) & 1;
      const int K  = ks * 32 + (j & 7) + ((j >> 3) * 16) + ((ln >> 4) * 8);
      const int n  = nt * 16 + (ln & 15);
      sWkF[idx] = (_Float16)Wk[K * HID + n];
      sWvF[idx] = (_Float16)Wv[K * HID + n];
    }
    for (int i = tid; i < HID; i += 256) {
      sFb[i] = PF(P_FIRE_B)[i];
      sBk[i] = PF(P_CA_BK)[i];
      sBv[i] = PF(P_CA_BV)[i];
    }
    for (int i = tid; i < 2 * HID; i += 256) sFw[i] = PF(P_FIRE_W)[i];
    for (int i = tid; i < ND * HID; i += 256) sQ[i] = ws_q[(size_t)b * ND * HID + i];
  }
  __syncthreads();

  const float* fpos  = PF(IN_FIREPOS) + (size_t)b * NF * 2;
  const float* gmask = out + OUT_MASK + (size_t)b * NF;
  float*       gfe   = out + OUT_FIRE_E + (size_t)b * NF * HID;

  // per-(q,head) flash state: each wave owns pairs wave*4 .. wave*4+3
  float mrun[4], lrun[4], oacc[4][DH];
  #pragma unroll
  for (int i = 0; i < 4; ++i) {
    mrun[i] = -1e30f; lrun[i] = 0.f;
    #pragma unroll
    for (int d = 0; d < DH; ++d) oacc[i][d] = 0.f;
  }

  // ---------- pass 1: fire_e + K/V projection (WMMA) + online softmax ----------
  for (int t = 0; t < NF / KT; ++t) {
    const int kt = t * KT;
    if (t + 1 < NF / KT)
      __builtin_prefetch(fpos + (size_t)(kt + KT) * 2, 0, 1);  // global_prefetch_b8

    // fire embedding tile; stream f32 to output, keep f16 in LDS
    #pragma unroll
    for (int j = 0; j < (KT * HID) / 256; ++j) {
      const int idx = j * 256 + tid;
      const int r = idx >> 6, c = idx & 63;
      const float x = fpos[(kt + r) * 2 + 0], y = fpos[(kt + r) * 2 + 1];
      const float e = fmaxf(x * sFw[c] + y * sFw[HID + c] + sFb[c], 0.f);
      gfe[(size_t)(kt + r) * HID + c] = e;
      sFe[idx] = (_Float16)e;
    }
    if (tid < KT) sMask[tid] = gmask[kt + tid];
    __syncthreads();

    // K proj on waves 0-3, V proj on waves 4-7 (uniform per-wave control flow)
    {
      const _Float16* WF  = (wave < 4) ? sWkF : sWvF;
      const float*   bias = (wave < 4) ? sBk : sBv;
      _Float16*      dst  = (wave < 4) ? sK16 : sV16;
      const int mt = wave & 3;
      #pragma unroll
      for (int nt = 0; nt < 4; ++nt) {
        v8f acc = {};
        #pragma unroll
        for (int ks = 0; ks < 2; ++ks) {
          v16h a  = load_a_frag(sFe, mt, ks, lane, HID);
          v16h bb = load_b_ready(WF, ks, nt, lane);
          acc = __builtin_amdgcn_wmma_f32_16x16x32_f16(false, a, false, bb,
                                                       (short)0, acc, false, false);
        }
        const int N = lane & 15, Mo = (lane >> 4) * 8;
        #pragma unroll
        for (int r = 0; r < 8; ++r)
          dst[(mt * 16 + Mo + r) * HID + nt * 16 + N] =
              (_Float16)(acc[r] + bias[nt * 16 + N]);
      }
    }
    __syncthreads();

    // online masked softmax + o accumulation (2 keys per lane)
    #pragma unroll
    for (int i = 0; i < 4; ++i) {
      const int pair = wave * 4 + i;
      const int hh = pair >> 3, q = pair & 7;
      // q row for this (q,head): uniform across the wave -> LDS broadcast reads
      float qr[DH];
      #pragma unroll
      for (int d = 0; d < DH; ++d) qr[d] = sQ[q * HID + hh * DH + d];

      float sv[2]; bool valid[2];
      #pragma unroll
      for (int j = 0; j < 2; ++j) {
        const int key = lane + j * 32;
        valid[j] = sMask[key] <= 0.5f;
        H16 kv;
        kv.v = *(const v16h*)&sK16[key * HID + hh * DH];   // 2x ds_load_b128
        float s = 0.f;
        #pragma unroll
        for (int d = 0; d < DH; ++d) s += qr[d] * (float)kv.h[d];
        sv[j] = valid[j] ? s * 0.25f : -1e30f;             // 1/sqrt(dh)
      }
      float tmax = fmaxf(sv[0], sv[1]);
      for (int off = 16; off; off >>= 1) tmax = fmaxf(tmax, __shfl_xor(tmax, off, 32));
      const float mnew = fmaxf(mrun[i], tmax);
      const float corr = __expf(mrun[i] - mnew);
      const float p0 = valid[0] ? __expf(sv[0] - mnew) : 0.f;
      const float p1 = valid[1] ? __expf(sv[1] - mnew) : 0.f;
      float ps = p0 + p1;
      for (int off = 16; off; off >>= 1) ps += __shfl_xor(ps, off, 32);
      lrun[i] = lrun[i] * corr + ps;

      H16 v0, v1;
      v0.v = *(const v16h*)&sV16[lane * HID + hh * DH];
      v1.v = *(const v16h*)&sV16[(lane + 32) * HID + hh * DH];
      #pragma unroll
      for (int d = 0; d < DH; ++d)
        oacc[i][d] = oacc[i][d] * corr + p0 * (float)v0.h[d] + p1 * (float)v1.h[d];
      mrun[i] = mnew;
    }
    __syncthreads();
  }

  // finalize o and stash (m,l) for pass 2
  #pragma unroll
  for (int i = 0; i < 4; ++i) {
    const int pair = wave * 4 + i;
    const int hh = pair >> 3, q = pair & 7;
    const float l = lrun[i];
    #pragma unroll
    for (int d = 0; d < DH; ++d) {
      float v = oacc[i][d];
      for (int off = 16; off; off >>= 1) v += __shfl_xor(v, off, 32);
      if (lane == 0) sO[q * HID + hh * DH + d] = v / l;
    }
    if (lane == 0) { sML[pair * 2] = mrun[i]; sML[pair * 2 + 1] = l; }
  }
  __syncthreads();

  // ---------- pass 2: recompute K proj, emit exact head-mean attention ---------
  float* gattn = out + OUT_ATTN + (size_t)b * ND * NF;
  for (int t = 0; t < NF / KT; ++t) {
    const int kt = t * KT;
    if (t + 1 < NF / KT)
      __builtin_prefetch(fpos + (size_t)(kt + KT) * 2, 0, 1);

    #pragma unroll
    for (int j = 0; j < (KT * HID) / 256; ++j) {
      const int idx = j * 256 + tid;
      const int r = idx >> 6, c = idx & 63;
      const float x = fpos[(kt + r) * 2 + 0], y = fpos[(kt + r) * 2 + 1];
      sFe[idx] = (_Float16)fmaxf(x * sFw[c] + y * sFw[HID + c] + sFb[c], 0.f);
    }
    if (tid < KT) sMask[tid] = gmask[kt + tid];
    __syncthreads();

    // K projection: all 8 waves, mt = wave&3, two N-tiles per wave
    {
      const int mt = wave & 3;
      const int nt0 = (wave >> 2) * 2;
      #pragma unroll
      for (int nt = nt0; nt < nt0 + 2; ++nt) {
        v8f acc = {};
        #pragma unroll
        for (int ks = 0; ks < 2; ++ks) {
          v16h a  = load_a_frag(sFe, mt, ks, lane, HID);
          v16h bb = load_b_ready(sWkF, ks, nt, lane);
          acc = __builtin_amdgcn_wmma_f32_16x16x32_f16(false, a, false, bb,
                                                       (short)0, acc, false, false);
        }
        const int N = lane & 15, Mo = (lane >> 4) * 8;
        #pragma unroll
        for (int r = 0; r < 8; ++r)
          sK16[(mt * 16 + Mo + r) * HID + nt * 16 + N] =
              (_Float16)(acc[r] + sBk[nt * 16 + N]);
      }
    }
    __syncthreads();

    // attn.mean over heads, normalized by pass-1 (m,l)
    #pragma unroll
    for (int j = 0; j < (ND * KT) / 256; ++j) {
      const int idx = j * 256 + tid;
      const int q = idx / KT, key = idx % KT;
      float a = 0.f;
      if (sMask[key] <= 0.5f) {
        #pragma unroll
        for (int hh = 0; hh < HEADS; ++hh) {
          const float m = sML[(hh * ND + q) * 2];
          const float l = sML[(hh * ND + q) * 2 + 1];
          H16 kv;
          kv.v = *(const v16h*)&sK16[key * HID + hh * DH];
          float s = 0.f;
          #pragma unroll
          for (int d = 0; d < DH; ++d)
            s += sQ[q * HID + hh * DH + d] * (float)kv.h[d];
          a += __expf(s * 0.25f - m) / l;
        }
        a *= 0.25f;  // mean over 4 heads
      }
      gattn[(size_t)q * NF + kt + key] = a;
    }
    __syncthreads();
  }

  // ---------- epilogue: o @ Wo, residual, LayerNorm -> drone_repr --------------
  if (tid < HID) {
    const int c = tid;
    const float* Wo = PF(P_CA_WO);
    for (int q = 0; q < ND; ++q) {
      float acc = PF(P_CA_BO)[c];
      for (int d = 0; d < HID; ++d) acc += sO[q * HID + d] * Wo[d * HID + c];
      sY[q * HID + c] = ws_demb[((size_t)b * ND + q) * HID + c] + acc;
    }
  }
  __syncthreads();
  if (tid < ND) {
    float m = 0.f;
    for (int c = 0; c < HID; ++c) m += sY[tid * HID + c];
    m /= HID;
    float v = 0.f;
    for (int c = 0; c < HID; ++c) { const float d3 = sY[tid * HID + c] - m; v += d3 * d3; }
    v /= HID;
    sStat[tid * 2] = m; sStat[tid * 2 + 1] = rsqrtf(v + 1e-5f);
  }
  __syncthreads();
  if (tid < HID) {
    for (int q = 0; q < ND; ++q)
      out[OUT_REPR + ((size_t)b * ND + q) * HID + tid] =
          (sY[q * HID + tid] - sStat[q * 2]) * sStat[q * 2 + 1] *
              PF(P_CA_LNG)[tid] + PF(P_CA_LNB)[tid];
  }
}

// ---------------- host launcher ------------------------------------------------
extern "C" void kernel_launch(void* const* d_in, const int* in_sizes, int n_in,
                              void* d_out, int out_size, void* d_ws, size_t ws_size,
                              hipStream_t stream) {
  (void)in_sizes; (void)out_size; (void)ws_size;
  PtrPack pp;
  for (int i = 0; i < N_IN && i < n_in; ++i) pp.p[i] = d_in[i];

  float* out = (float*)d_out;
  float* ws  = (float*)d_ws;
  float* ws_demb = ws;                              // B*ND*HID
  float* ws_q    = ws + (size_t)BATCH * ND * HID;   // B*ND*HID

  mask_kernel<<<BATCH, 256, 0, stream>>>((const unsigned char*)d_in[IN_FIREMASK], out);
  drone_encoder<<<BATCH, 64, 0, stream>>>(pp, ws_demb, ws_q);
  fire_cross_attn<<<BATCH, 256, 0, stream>>>(pp, ws_demb, ws_q, out);
}
